// GlobalPointer_2104533975796
// MI455X (gfx1250) — matrix operand order repaired and checked by reading, not compiled, not run on hardware
//
#include <hip/hip_runtime.h>
#include <hip/hip_bf16.h>

// ---------------------------------------------------------------------------
// GlobalPointer head for MI455X (gfx1250): bf16 WMMA GEMMs, RoPE fused into
// the projection epilogue (register-level pairing), fused mask epilogue.
// B=8, S=1024, H=768, E=9, D=64, N=E*2*D=1152.
// ---------------------------------------------------------------------------

#define Bsz   8
#define Ssz   1024
#define Hsz   768
#define Esz   9
#define Dsz   64
#define Nsz   (Esz * 2 * Dsz)      // 1152
#define Msz   (Bsz * Ssz)          // 8192

typedef __attribute__((ext_vector_type(16))) __bf16 v16bf;
typedef __attribute__((ext_vector_type(8)))  float  v8f;

union Frag16 {
    v16bf v;
    uint4 q[2];
};

// ---- WMMA fragment loaders (bf16, ISA 7.12.2 layouts) ----------------------
// A matrix 16x32 (MxK), row-major source with leading dim `ld`:
//   lanes 0-15 : row M=lane,    K = 0..7 (vgpr0-3), 16..23 (vgpr4-7)
//   lanes 16-31: row M=lane-16, K = 8..15,          24..31
__device__ __forceinline__ v16bf load_a_frag(const __bf16* __restrict__ src,
                                             int ld, int row0, int k0) {
    const int lane = threadIdx.x & 31;
    const int r  = row0 + (lane & 15);
    const int kk = k0 + ((lane >> 4) << 3);          // +0 or +8
    const __bf16* base = src + (size_t)r * ld + kk;
    Frag16 f;
    f.q[0] = *reinterpret_cast<const uint4*>(base);       // K=kk..kk+7
    f.q[1] = *reinterpret_cast<const uint4*>(base + 16);  // K=kk+16..kk+23
    return f.v;
}

// B matrix 32x16 (KxN); source stores each column n contiguously over k
// (row n of the transposed operand), leading dim `ld`:
//   lanes 0-15 : col N=lane,    K = 0..7, 16..23
//   lanes 16-31: col N=lane-16, K = 8..15, 24..31
__device__ __forceinline__ v16bf load_b_frag(const __bf16* __restrict__ src,
                                             int ld, int col0, int k0) {
    const int lane = threadIdx.x & 31;
    const int n  = col0 + (lane & 15);
    const int kk = k0 + ((lane >> 4) << 3);          // +0 or +8
    const __bf16* base = src + (size_t)n * ld + kk;
    Frag16 f;
    f.q[0] = *reinterpret_cast<const uint4*>(base);
    f.q[1] = *reinterpret_cast<const uint4*>(base + 16);
    return f.v;
}

// ---- fp32 -> bf16 conversion prepass ---------------------------------------
__global__ void cvt_f32_to_bf16(const float* __restrict__ in,
                                __bf16* __restrict__ out, int n4) {
    int i = blockIdx.x * blockDim.x + threadIdx.x;
    if (i >= n4) return;
    float4 v = reinterpret_cast<const float4*>(in)[i];
    __bf16 t[4] = { (__bf16)v.x, (__bf16)v.y, (__bf16)v.z, (__bf16)v.w };
    reinterpret_cast<uint2*>(out)[i] = *reinterpret_cast<uint2*>(t);
}

// ---- Kernel 1: projection GEMM + bias + RoPE, direct bf16 q/k output -------
// Y = X @ W^T + b ; then q' = q*cos + rot(q)*sin per (s,d); store [b,e,s,d].
// Tile 128(M) x 128(N) per workgroup. 8 waves as 4(M) x 2(N); each wave owns
// 32 rows x 64 cols = acc[2][4] 16x16 frags. Since tileN = e*128 and each
// wave covers one full 64-wide q-half or k-half, the RoPE partner of frag j
// is frag j+2 at the same lane/slot -> pure register pairing.
__global__ __launch_bounds__(256) void proj_rope_bf16(
    const __bf16* __restrict__ X,   // [M, 768] row-major
    const __bf16* __restrict__ W,   // [1152, 768] row-major (== B columns)
    const float*  __restrict__ bias,
    __bf16* __restrict__ qw,        // [B*E][S][D]
    __bf16* __restrict__ kw)        // [B*E][S][D]
{
    const int wave = threadIdx.x >> 5;
    const int wm = wave >> 1;            // 0..3
    const int wn = wave & 1;             // 0..1
    const int e  = blockIdx.y;           // tileN = e*128
    const int rowBase = blockIdx.x * 128 + wm * 32;
    const int colBase = e * 128 + wn * 64;

    v8f acc[2][4];
#pragma unroll
    for (int i = 0; i < 2; ++i)
#pragma unroll
        for (int j = 0; j < 4; ++j) acc[i][j] = v8f{};

    for (int kb = 0; kb < Hsz; kb += 32) {
        // speculative next-K prefetch (overshoot lands in mapped workspace)
        __builtin_prefetch(X + (size_t)(rowBase + (threadIdx.x & 31)) * Hsz + kb + 32, 0, 1);
        __builtin_prefetch(W + (size_t)(colBase + (threadIdx.x & 31)) * Hsz + kb + 32, 0, 1);

        v16bf a[2], bf[4];
#pragma unroll
        for (int i = 0; i < 2; ++i)
            a[i] = load_a_frag(X, Hsz, rowBase + i * 16, kb);
#pragma unroll
        for (int j = 0; j < 4; ++j)
            bf[j] = load_b_frag(W, Hsz, colBase + j * 16, kb);
#pragma unroll
        for (int i = 0; i < 2; ++i)
#pragma unroll
            for (int j = 0; j < 4; ++j)
                acc[i][j] = __builtin_amdgcn_wmma_f32_16x16x32_bf16(
                    false, a[i], false, bf[j], (short)0, acc[i][j], false, false);
    }

    // ---- fused epilogue: bias + RoPE + bf16 store --------------------------
    const int lane = threadIdx.x & 31;
    const int cl = lane & 15;
    const int rowHalf = (lane >> 4) << 3;            // 0 or 8
    __bf16* __restrict__ dst = (wn == 1) ? kw : qw;  // wave-uniform

    // per-frag constants: bias and inverse frequency (depend on d only)
    float bv[4], invf[4];
#pragma unroll
    for (int j = 0; j < 4; ++j) {
        const int d = 16 * j + cl;                   // 0..63 within the half
        bv[j]   = bias[colBase + d];
        // inv_freq[d>>1] = 10000^(-2*(d>>1)/64) = exp(-(d>>1)*0.2878231366)
        invf[j] = __expf(-0.28782313662425572f * (float)(d >> 1));
    }

#pragma unroll
    for (int i = 0; i < 2; ++i) {
#pragma unroll
        for (int r = 0; r < 8; ++r) {
            const int m = rowBase + 16 * i + rowHalf + r;
            const int b = m >> 10;                   // m / S
            const int s = m & (Ssz - 1);             // m % S
            const float fs = (float)s;
            const size_t obase = (size_t)((b * Esz + e) * Ssz + s) * Dsz;
#pragma unroll
            for (int jl = 0; jl < 2; ++jl) {
                const int dlo = 16 * jl + cl;        // 0..31
                const float xlo = acc[i][jl][r]     + bv[jl];
                const float xhi = acc[i][jl + 2][r] + bv[jl + 2];
                float snl, csl, snh, csh;
                __sincosf(fs * invf[jl],     &snl, &csl);
                __sincosf(fs * invf[jl + 2], &snh, &csh);
                // d<32:  x[d]*cos - x[d+32]*sin ; d>=32: x[d]*cos + x[d-32]*sin
                dst[obase + dlo]      = (__bf16)(xlo * csl - xhi * snl);
                dst[obase + dlo + 32] = (__bf16)(xhi * csh + xlo * snh);
            }
        }
    }
}

// ---- Kernel 2: logits[b,e,m,n] = (qw . kw)/8 with pad + triangular mask ----
// 72 batched GEMMs 1024x1024x64; 128x128 tile/workgroup, fused epilogue.
__global__ __launch_bounds__(256) void logits_gemm_bf16(
    const __bf16* __restrict__ qw,   // [B*E][S][D]
    const __bf16* __restrict__ kw,   // [B*E][S][D]
    const float*  __restrict__ mask, // [B][S]
    float* __restrict__ out)         // [B*E][S][S]
{
    const int be = blockIdx.z;
    const int b  = be / Esz;
    const __bf16* Q = qw + (size_t)be * Ssz * Dsz;
    const __bf16* K = kw + (size_t)be * Ssz * Dsz;

    const int wave = threadIdx.x >> 5;
    const int wm = wave >> 2;
    const int wn = wave & 3;
    const int rowBase = blockIdx.x * 128 + wm * 64;
    const int colBase = blockIdx.y * 128 + wn * 32;

    v8f acc[4][2];
#pragma unroll
    for (int i = 0; i < 4; ++i)
#pragma unroll
        for (int j = 0; j < 2; ++j) acc[i][j] = v8f{};

#pragma unroll
    for (int kb = 0; kb < Dsz; kb += 32) {
        v16bf a[4], bf[2];
#pragma unroll
        for (int i = 0; i < 4; ++i)
            a[i] = load_a_frag(Q, Dsz, rowBase + i * 16, kb);
#pragma unroll
        for (int j = 0; j < 2; ++j)
            bf[j] = load_b_frag(K, Dsz, colBase + j * 16, kb);
#pragma unroll
        for (int i = 0; i < 4; ++i)
#pragma unroll
            for (int j = 0; j < 2; ++j)
                acc[i][j] = __builtin_amdgcn_wmma_f32_16x16x32_bf16(
                    false, a[i], false, bf[j], (short)0, acc[i][j], false, false);
    }

    const int lane = threadIdx.x & 31;
    const int rowHalf = (lane >> 4) << 3;
    const float scale = 0.125f;                 // 1/sqrt(64)
#pragma unroll
    for (int i = 0; i < 4; ++i)
#pragma unroll
        for (int j = 0; j < 2; ++j) {
            const int n = colBase + j * 16 + (lane & 15);
            const float pad = mask[b * Ssz + n];
            const float padBias = -(1.0f - pad) * 10000.0f;
#pragma unroll
            for (int r = 0; r < 8; ++r) {
                const int m = rowBase + i * 16 + rowHalf + r;
                float v = acc[i][j][r] * scale * pad + padBias;
                if (n < m) v -= 10000.0f;       // (1 - triu) mask
                out[(size_t)(be * Ssz + m) * Ssz + n] = v;
            }
        }
}

// ---------------------------------------------------------------------------
extern "C" void kernel_launch(void* const* d_in, const int* in_sizes, int n_in,
                              void* d_out, int out_size, void* d_ws, size_t ws_size,
                              hipStream_t stream) {
    const float* X    = (const float*)d_in[0];   // [8,1024,768]
    const float* mask = (const float*)d_in[1];   // [8,1024]
    const float* W    = (const float*)d_in[2];   // [1152,768]
    const float* bias = (const float*)d_in[3];   // [1152]
    float* out = (float*)d_out;                  // [8,9,1024,1024]

    char* ws = (char*)d_ws;
    // workspace layout (bytes), all 16B aligned (~33.2 MB total):
    __bf16* Xb = (__bf16*)(ws);                  // 8192*768*2    = 12,582,912
    __bf16* Wb = (__bf16*)(ws + 12582912);       // 1152*768*2    =  1,769,472
    __bf16* qw = (__bf16*)(ws + 14352384);       // 8*9*1024*64*2 =  9,437,184
    __bf16* kw = (__bf16*)(ws + 23789568);       // 8*9*1024*64*2 =  9,437,184

    // 1) convert X and W to bf16 (vectorized x4)
    {
        const int n4x = (Msz * Hsz) / 4;              // 1,572,864
        cvt_f32_to_bf16<<<(n4x + 255) / 256, 256, 0, stream>>>(X, Xb, n4x);
        const int n4w = (Nsz * Hsz) / 4;              // 221,184
        cvt_f32_to_bf16<<<(n4w + 255) / 256, 256, 0, stream>>>(W, Wb, n4w);
    }

    // 2) projection GEMM + bias + RoPE fused -> qw/kw bf16
    {
        dim3 grid(Msz / 128, Nsz / 128, 1);           // (64, 9)
        proj_rope_bf16<<<grid, 256, 0, stream>>>(Xb, Wb, bias, qw, kw);
    }

    // 3) batched logits GEMM with fused scale/pad/tri masks
    {
        dim3 grid(Ssz / 128, Ssz / 128, Bsz * Esz);   // (8, 8, 72)
        logits_gemm_bf16<<<grid, 256, 0, stream>>>(qw, kw, mask, out);
    }
}
